// SimplifiedGNNEncoder_7928509628561
// MI455X (gfx1250) — compile-verified, matrix operand order checked by reading, not run on hardware
//
#include <hip/hip_runtime.h>
#include <hip/hip_bf16.h>

// ---------------------------------------------------------------------------
// SimplifiedGNNEncoder for MI455X (gfx1250, wave32).
// Memory-bound (~5 GB traffic vs ~25 GFLOP). GEMMs use
// v_wmma_f32_16x16x32_bf16 with bf16 operands pre-staged in memory so the
// WMMA kernels are pure b128-load / wmma / store (no per-element cvt VALU).
// ---------------------------------------------------------------------------

typedef __attribute__((ext_vector_type(16))) __bf16 v16bf;
typedef __attribute__((ext_vector_type(8)))  __bf16 v8bf;
typedef __attribute__((ext_vector_type(4)))  __bf16 v4bf;
typedef __attribute__((ext_vector_type(8)))  float  v8f;

#define FD 64           // feature dim
#define BN_EPS 1e-5f

// float -> bf16, round-to-nearest-even (used only in streaming kernels)
__device__ __forceinline__ __bf16 f2bf(float f) {
    unsigned u = __builtin_bit_cast(unsigned, f);
    unsigned r = (u + 0x7FFFu + ((u >> 16) & 1u)) >> 16;
    return __builtin_bit_cast(__bf16, (unsigned short)r);
}

__device__ __forceinline__ float silu(float v) { return v / (1.0f + __expf(-v)); }

// Load one 16-element bf16 WMMA fragment from a row-major bf16 row.
// ISA 16-bit A/B layout: element e <-> K = k0 + 16*(e>>3) + 8*half + (e&7)
// -> two contiguous 8-bf16 (16 B) runs: exactly two global_load_b128.
__device__ __forceinline__ v16bf load_frag(const __bf16* __restrict__ row,
                                           int k0, int half) {
    v8bf lo = *reinterpret_cast<const v8bf*>(row + k0 + 8 * half);
    v8bf hi = *reinterpret_cast<const v8bf*>(row + k0 + 16 + 8 * half);
    return __builtin_shufflevector(lo, hi, 0, 1, 2, 3, 4, 5, 6, 7,
                                   8, 9, 10, 11, 12, 13, 14, 15);
}

// One wave computes a 16-row x 64-col stripe of  out = A @ W^T + bias.
__device__ __forceinline__ void gemm_tile_16x64(const __bf16* __restrict__ A,
                                                int m0,
                                                const __bf16* __restrict__ W,
                                                const float* __restrict__ bias,
                                                float* __restrict__ out,
                                                int lane) {
    const int half = lane >> 4, r = lane & 15;
    const __bf16* arow = A + (size_t)(m0 + r) * FD;
    v16bf a0 = load_frag(arow, 0, half);
    v16bf a1 = load_frag(arow, 32, half);
#pragma unroll
    for (int nt = 0; nt < 4; ++nt) {
        const __bf16* brow = W + (size_t)(nt * 16 + r) * FD;   // col nt*16+r of B
        v16bf b0 = load_frag(brow, 0, half);
        v16bf b1 = load_frag(brow, 32, half);
        v8f acc = {0.f, 0.f, 0.f, 0.f, 0.f, 0.f, 0.f, 0.f};
        acc = __builtin_amdgcn_wmma_f32_16x16x32_bf16(false, a0, false, b0,
                                                      (short)0, acc, false, false);
        acc = __builtin_amdgcn_wmma_f32_16x16x32_bf16(false, a1, false, b1,
                                                      (short)0, acc, false, false);
        const float bb = bias[nt * 16 + r];
#pragma unroll
        for (int i = 0; i < 8; ++i)
            out[(size_t)(m0 + i + 8 * half) * FD + nt * 16 + r] = acc[i] + bb;
    }
}

// 4 node linears: xk = h @ Wv[k]^T + bv[k]
__global__ void node_gemm_kernel(const __bf16* __restrict__ hb,
                                 const __bf16* __restrict__ Wvb_l,
                                 const float* __restrict__ bv_l,
                                 float* __restrict__ xout, int nrows) {
    const int wid  = threadIdx.x >> 5;
    const int lane = threadIdx.x & 31;
    const int gw = blockIdx.x * (blockDim.x >> 5) + wid;
    const int ntiles = nrows >> 4;
    if (gw >= ntiles * 4) return;
    const int k = gw & 3, mt = gw >> 2;
    gemm_tile_16x64(hb, mt * 16, Wvb_l + k * FD * FD, bv_l + k * FD,
                    xout + (size_t)k * (size_t)nrows * FD, lane);
}

// Edge linear: te = w @ We^T + be
__global__ void edge_gemm_kernel(const __bf16* __restrict__ wb,
                                 const __bf16* __restrict__ Web_l,
                                 const float* __restrict__ be_l,
                                 float* __restrict__ te, int erows) {
    const int wid  = threadIdx.x >> 5;
    const int lane = threadIdx.x & 31;
    const int gw = blockIdx.x * (blockDim.x >> 5) + wid;
    if (gw >= (erows >> 4)) return;
    gemm_tile_16x64(wb, gw * 16, Web_l, be_l, te, lane);
}

// f32 -> bf16 bulk convert (weights), float4-vectorized
__global__ void cvt_bf16_kernel(const float* __restrict__ in,
                                __bf16* __restrict__ out, int n4) {
    int idx = blockIdx.x * blockDim.x + threadIdx.x;
    if (idx >= n4) return;
    float4 v = reinterpret_cast<const float4*>(in)[idx];
    v4bf o; o[0] = f2bf(v.x); o[1] = f2bf(v.y); o[2] = f2bf(v.z); o[3] = f2bf(v.w);
    reinterpret_cast<v4bf*>(out)[idx] = o;
}

// out = silu(in), with fused bf16 mirror write
__global__ void silu_kernel(const float* __restrict__ in,
                            float* __restrict__ out,
                            __bf16* __restrict__ outb, int n4) {
    int idx = blockIdx.x * blockDim.x + threadIdx.x;
    if (idx >= n4) return;
    float4 v = reinterpret_cast<const float4*>(in)[idx];
    v.x = silu(v.x); v.y = silu(v.y); v.z = silu(v.z); v.w = silu(v.w);
    reinterpret_cast<float4*>(out)[idx] = v;
    v4bf o; o[0] = f2bf(v.x); o[1] = f2bf(v.y); o[2] = f2bf(v.z); o[3] = f2bf(v.w);
    reinterpret_cast<v4bf*>(outb)[idx] = o;
}

__global__ void zero_kernel(float* __restrict__ p, int n) {
    int i = blockIdx.x * blockDim.x + threadIdx.x;
    if (i < n) p[i] = 0.0f;
}

// msg = sigmoid(w) * x2[dst]; scatter-mean numerator + count onto src
__global__ void scatter_kernel(const float* __restrict__ w,
                               const float* __restrict__ x2,
                               const int* __restrict__ src,
                               const int* __restrict__ dst,
                               float* __restrict__ agg,
                               float* __restrict__ cnt, int E) {
    int idx = blockIdx.x * blockDim.x + threadIdx.x;   // E*16 float4 slots
    if (idx >= E * 16) return;
    int e = idx >> 4, q = idx & 15;
    int s = src[e], d = dst[e];
    float4 wv = reinterpret_cast<const float4*>(w)[idx];
    float4 xv = reinterpret_cast<const float4*>(x2)[(size_t)d * 16 + q];
    float m0 = xv.x / (1.0f + __expf(-wv.x));
    float m1 = xv.y / (1.0f + __expf(-wv.y));
    float m2 = xv.z / (1.0f + __expf(-wv.z));
    float m3 = xv.w / (1.0f + __expf(-wv.w));
    float* a = agg + (size_t)s * FD + q * 4;
    atomicAdd(a + 0, m0); atomicAdd(a + 1, m1);
    atomicAdd(a + 2, m2); atomicAdd(a + 3, m3);
    if (q == 0) atomicAdd(cnt + s, 1.0f);
}

// tn = x1 + agg / max(cnt, 1)
__global__ void tn_kernel(const float* __restrict__ x1,
                          const float* __restrict__ agg,
                          const float* __restrict__ cnt,
                          float* __restrict__ tn, int N) {
    int idx = blockIdx.x * blockDim.x + threadIdx.x;   // N*16 float4 slots
    if (idx >= N * 16) return;
    int n = idx >> 4;
    float c = fmaxf(cnt[n], 1.0f);
    float4 a = reinterpret_cast<const float4*>(agg)[idx];
    float4 v = reinterpret_cast<const float4*>(x1)[idx];
    v.x += a.x / c; v.y += a.y / c; v.z += a.z / c; v.w += a.w / c;
    reinterpret_cast<float4*>(tn)[idx] = v;
}

// te += x3[src] + x4[dst]
__global__ void gather_add_kernel(float* __restrict__ te,
                                  const float* __restrict__ x3,
                                  const float* __restrict__ x4,
                                  const int* __restrict__ src,
                                  const int* __restrict__ dst, int E) {
    int idx = blockIdx.x * blockDim.x + threadIdx.x;   // E*16 float4 slots
    if (idx >= E * 16) return;
    int e = idx >> 4, q = idx & 15;
    float4 t = reinterpret_cast<const float4*>(te)[idx];
    float4 a = reinterpret_cast<const float4*>(x3)[(size_t)src[e] * 16 + q];
    float4 b = reinterpret_cast<const float4*>(x4)[(size_t)dst[e] * 16 + q];
    t.x += a.x + b.x; t.y += a.y + b.y; t.z += a.z + b.z; t.w += a.w + b.w;
    reinterpret_cast<float4*>(te)[idx] = t;
}

// per-feature sum / sumsq (64 threads = 64 features; rows coalesced per block)
__global__ void bn_stats_kernel(const float* __restrict__ t, int rows,
                                float* __restrict__ sum,
                                float* __restrict__ sumsq) {
    int f = threadIdx.x;                      // blockDim.x == 64
    float s = 0.f, q = 0.f;
    for (int r = blockIdx.x; r < rows; r += gridDim.x) {
        float v = t[(size_t)r * FD + f];
        s += v; q += v * v;
    }
    atomicAdd(sum + f, s);
    atomicAdd(sumsq + f, q);
}

__global__ void bn_finalize_kernel(const float* __restrict__ sum,
                                   const float* __restrict__ sumsq, int rows,
                                   float* __restrict__ mean,
                                   float* __restrict__ inv) {
    int f = threadIdx.x;                      // 1 block, 64 threads
    float inv_r = 1.0f / (float)rows;
    float mu  = sum[f] * inv_r;
    float var = sumsq[f] * inv_r - mu * mu;   // biased variance
    mean[f] = mu;
    inv[f]  = rsqrtf(var + BN_EPS);
}

// hw += silu(gamma*(t-mean)*invstd + beta); fused bf16 mirror write
__global__ void apply_kernel(float* __restrict__ hw,
                             __bf16* __restrict__ hwb,
                             const float* __restrict__ t,
                             const float* __restrict__ mean,
                             const float* __restrict__ inv,
                             const float* __restrict__ g,
                             const float* __restrict__ b, int n4) {
    int idx = blockIdx.x * blockDim.x + threadIdx.x;   // float4 slots
    if (idx >= n4) return;
    int f = (idx & 15) * 4;
    float4 tv = reinterpret_cast<const float4*>(t)[idx];
    float4 hv = reinterpret_cast<const float4*>(hw)[idx];
    float z0 = g[f + 0] * (tv.x - mean[f + 0]) * inv[f + 0] + b[f + 0];
    float z1 = g[f + 1] * (tv.y - mean[f + 1]) * inv[f + 1] + b[f + 1];
    float z2 = g[f + 2] * (tv.z - mean[f + 2]) * inv[f + 2] + b[f + 2];
    float z3 = g[f + 3] * (tv.w - mean[f + 3]) * inv[f + 3] + b[f + 3];
    hv.x += silu(z0); hv.y += silu(z1); hv.z += silu(z2); hv.w += silu(z3);
    reinterpret_cast<float4*>(hw)[idx] = hv;
    v4bf o; o[0] = f2bf(hv.x); o[1] = f2bf(hv.y); o[2] = f2bf(hv.z); o[3] = f2bf(hv.w);
    reinterpret_cast<v4bf*>(hwb)[idx] = o;
}

extern "C" void kernel_launch(void* const* d_in, const int* in_sizes, int n_in,
                              void* d_out, int out_size, void* d_ws, size_t ws_size,
                              hipStream_t stream) {
    const float* x    = (const float*)d_in[0];
    const int*   ei   = (const int*)  d_in[1];
    const float* ea   = (const float*)d_in[2];
    const float* Wv   = (const float*)d_in[3];
    const float* bv   = (const float*)d_in[4];
    const float* We   = (const float*)d_in[5];
    const float* be   = (const float*)d_in[6];
    const float* g_n  = (const float*)d_in[7];
    const float* b_n  = (const float*)d_in[8];
    const float* g_e  = (const float*)d_in[9];
    const float* b_e  = (const float*)d_in[10];

    const int N = in_sizes[0] / FD;       // 50000
    const int E = in_sizes[2] / FD;       // 800000
    const int L = in_sizes[6] / FD;       // 3
    const int* src = ei;
    const int* dst = ei + E;

    const size_t ND = (size_t)N * FD, ED = (size_t)E * FD;

    // d_out doubles as the live h / w buffers
    float* h = (float*)d_out;             // [N, 64]
    float* w = h + ND;                    // [E, 64]

    // ---- workspace layout: f32 region, then bf16 region ----
    float* ws  = (float*)d_ws;
    float* x1  = ws;                      // 4 x [N,64] node linears
    float* x2  = x1 + ND;
    float* x3  = x2 + ND;
    float* x4  = x3 + ND;
    float* agg = x4 + ND;                 // [N,64]
    float* cnt = agg + ND;                // [N]
    float* tn  = cnt + N;                 // [N,64] pre-BN node
    float* te  = tn + ND;                 // [E,64] pre-BN edge
    float* st  = te + ED;                 // 512 floats of stats
    float *nsum = st,       *nsq = st + 64,  *nmean = st + 128, *ninv = st + 192;
    float *esum = st + 256, *esq = st + 320, *emean = st + 384, *einv = st + 448;

    __bf16* hb  = (__bf16*)(st + 512);    // [N,64] bf16 mirror of h
    __bf16* wb  = hb + ND;                // [E,64] bf16 mirror of w
    __bf16* Wvb = wb + ED;                // [L,4,64,64] bf16 weights
    __bf16* Web = Wvb + (size_t)L * 4 * FD * FD;   // [L,64,64]

    auto blk = [](size_t n) { return (unsigned)((n + 255) / 256); };

    // weights -> bf16 (once per launch)
    const int wv4 = L * 4 * FD * FD / 4, we4 = L * FD * FD / 4;
    cvt_bf16_kernel<<<blk((size_t)wv4), 256, 0, stream>>>(Wv, Wvb, wv4);
    cvt_bf16_kernel<<<blk((size_t)we4), 256, 0, stream>>>(We, Web, we4);

    // h = silu(x), w = silu(edge_attr)  (+ bf16 mirrors)
    silu_kernel<<<blk(ND / 4), 256, 0, stream>>>(x, h, hb, (int)(ND / 4));
    silu_kernel<<<blk(ED / 4), 256, 0, stream>>>(ea, w, wb, (int)(ED / 4));

    const int nblk_node = ((N / 16) * 4 + 7) / 8;   // 8 waves / 256-thr block
    const int nblk_edge = ((E / 16) + 7) / 8;

    for (int l = 0; l < L; ++l) {
        const __bf16* Wvbl = Wvb + (size_t)l * 4 * FD * FD;
        const float*  bvl  = bv + (size_t)l * 4 * FD;
        const __bf16* Webl = Web + (size_t)l * FD * FD;
        const float*  bel  = be + (size_t)l * FD;

        // x1..x4 = h @ Wv[k]^T + bv[k]   (WMMA, bf16 operands)
        node_gemm_kernel<<<nblk_node, 256, 0, stream>>>(hb, Wvbl, bvl, x1, N);

        zero_kernel<<<blk(ND), 256, 0, stream>>>(agg, (int)ND);
        zero_kernel<<<blk((size_t)N), 256, 0, stream>>>(cnt, N);
        zero_kernel<<<1, 512, 0, stream>>>(st, 512);

        // scatter-mean of sigmoid(w) * x2[dst] onto src
        scatter_kernel<<<blk((size_t)E * 16), 256, 0, stream>>>(w, x2, src, dst,
                                                                agg, cnt, E);
        tn_kernel<<<blk((size_t)N * 16), 256, 0, stream>>>(x1, agg, cnt, tn, N);

        // node BN stats
        bn_stats_kernel<<<512, 64, 0, stream>>>(tn, N, nsum, nsq);
        bn_finalize_kernel<<<1, 64, 0, stream>>>(nsum, nsq, N, nmean, ninv);

        // te = w @ We^T + be  (WMMA), then += x3[src] + x4[dst]
        edge_gemm_kernel<<<nblk_edge, 256, 0, stream>>>(wb, Webl, bel, te, E);
        gather_add_kernel<<<blk((size_t)E * 16), 256, 0, stream>>>(te, x3, x4,
                                                                   src, dst, E);

        // h += silu(bn(tn))  (x3/x4 already captured old h; updates hb too)
        apply_kernel<<<blk(ND / 4), 256, 0, stream>>>(h, hb, tn, nmean, ninv,
                                                      g_n + l * FD, b_n + l * FD,
                                                      (int)(ND / 4));

        // edge BN stats + apply: w += silu(bn(te))  (updates wb for next layer)
        bn_stats_kernel<<<1024, 64, 0, stream>>>(te, E, esum, esq);
        bn_finalize_kernel<<<1, 64, 0, stream>>>(esum, esq, E, emean, einv);
        apply_kernel<<<blk(ED / 4), 256, 0, stream>>>(w, wb, te, emean, einv,
                                                      g_e + l * FD, b_e + l * FD,
                                                      (int)(ED / 4));
    }
    (void)n_in; (void)out_size; (void)ws_size;
}
